// TritonConv2dInt8_31782757990428
// MI455X (gfx1250) — compile-verified, matrix operand order checked by reading, not run on hardware
//
#include <hip/hip_runtime.h>
#include <stdint.h>

// ---------------------------------------------------------------------------
// Int8 quantized 3x3 conv (pad 1) as implicit GEMM on CDNA5 WMMA IU8.
//   GEMM: A = w_q [256 x 1152], B = im2col(x_q) [1152 x 100352], C int32.
//   K ordering = (r, s, c) so each 64-wide WMMA K-slab is a contiguous
//   channel run of the NHWC-quantized activations.
// ---------------------------------------------------------------------------

typedef __attribute__((ext_vector_type(8))) int v8i;

#define QMAX 127.0f

constexpr int NB = 32, C = 128, H = 56, W = 56, OC = 256;
constexpr int HWc   = H * W;          // 3136
constexpr int Mpix  = NB * HWc;       // 100352  (divisible by 128)
constexpr int Kdim  = C * 9;          // 1152
constexpr int KITER = Kdim / 64;      // 18

// ---------------------------------------------------------------- scales ----
__global__ void zero_scales_kernel(float* wsf) {
    if (threadIdx.x < 2) wsf[threadIdx.x] = 0.0f;
}

__global__ void absmax_kernel(const float4* __restrict__ src, long n4,
                              float* __restrict__ dst) {
    __shared__ float red[256];
    float m = 0.0f;
    for (long i = (long)blockIdx.x * blockDim.x + threadIdx.x; i < n4;
         i += (long)gridDim.x * blockDim.x) {
        float4 v = src[i];
        m = fmaxf(m, fmaxf(fmaxf(fabsf(v.x), fabsf(v.y)),
                           fmaxf(fabsf(v.z), fabsf(v.w))));
    }
    red[threadIdx.x] = m;
    __syncthreads();
    for (int s = 128; s > 0; s >>= 1) {
        if ((int)threadIdx.x < s)
            red[threadIdx.x] = fmaxf(red[threadIdx.x], red[threadIdx.x + s]);
        __syncthreads();
    }
    // abs values are non-negative: uint compare == float compare. Exact+deterministic.
    if (threadIdx.x == 0) atomicMax((unsigned int*)dst, __float_as_uint(red[0]));
}

__device__ __forceinline__ signed char quant1(float v, float inv) {
    float q = rintf(v * inv);                 // round-half-even, matches jnp.round
    q = fminf(fmaxf(q, -QMAX), QMAX);
    return (signed char)(int)q;
}

// x: fp32 NCHW -> x_q: int8 NHWC (4 channels per thread, packed dword store)
__global__ void quantize_x_kernel(const float* __restrict__ x,
                                  const float* __restrict__ wsf,
                                  char4* __restrict__ xq) {
    const long total = (long)NB * HWc * (C / 4);
    long tid = (long)blockIdx.x * blockDim.x + threadIdx.x;
    if (tid >= total) return;
    const float sx  = fmaxf(wsf[0] / QMAX, 1e-8f);
    const float inv = 1.0f / sx;
    const int  c4 = (int)(tid & 31);          // C/4 = 32
    const long p  = tid >> 5;                 // n*HW + h*W + w
    const long base = ((p / HWc) * C) * HWc + (p % HWc); // x[n][0][h][w]
    const int  c = c4 * 4;
    char4 q;
    q.x = quant1(x[base + (long)(c + 0) * HWc], inv);
    q.y = quant1(x[base + (long)(c + 1) * HWc], inv);
    q.z = quant1(x[base + (long)(c + 2) * HWc], inv);
    q.w = quant1(x[base + (long)(c + 3) * HWc], inv);
    xq[p * 32 + c4] = q;
}

// w: fp32 OIHW -> w_q: int8 [oc][(r*3+s)*128 + c]
__global__ void quantize_w_kernel(const float* __restrict__ wgt,
                                  const float* __restrict__ wsf,
                                  char4* __restrict__ wq) {
    const int total = OC * 9 * (C / 4);       // 73728
    int tid = blockIdx.x * blockDim.x + threadIdx.x;
    if (tid >= total) return;
    const float sw  = fmaxf(wsf[1] / QMAX, 1e-8f);
    const float inv = 1.0f / sw;
    const int c4 = tid & 31;
    const int rs = (tid >> 5) % 9;
    const int o  = tid / 288;
    const long base = (long)o * C * 9 + rs;   // wgt[o][c][r][s] = o*1152 + c*9 + rs
    const int  c = c4 * 4;
    char4 q;
    q.x = quant1(wgt[base + (long)(c + 0) * 9], inv);
    q.y = quant1(wgt[base + (long)(c + 1) * 9], inv);
    q.z = quant1(wgt[base + (long)(c + 2) * 9], inv);
    q.w = quant1(wgt[base + (long)(c + 3) * 9], inv);
    wq[o * 288 + rs * 32 + c4] = q;           // char4 index
}

// ------------------------------------------------------------- main GEMM ----
// Block: 256 threads (8 wave32). Tile: 128 oc x 128 pixels, K-loop 18x64.
// LDS rows padded to 80 B -> ds_load_b64 fragment reads are bank-conflict free.
__global__ __launch_bounds__(256)
void conv_gemm_i8_wmma(const signed char* __restrict__ xq,
                       const signed char* __restrict__ wq,
                       const float* __restrict__ bias,
                       const float* __restrict__ wsf,
                       float* __restrict__ out) {
    __shared__ __align__(16) signed char shW[2][128][80];  // weight tile [oc][K64]
    __shared__ __align__(16) signed char shA[2][128][80];  // activation tile [pix][K64]

    const int tid     = threadIdx.x;
    const int pixBase = blockIdx.x * 128;
    const int ocBase  = blockIdx.y * 128;

    // --- cooperative loaders: 2 threads per 64B row, 32B (2x uint4) each ---
    const int lrowIdx = tid >> 1;
    const int lchunk  = (tid & 1) * 32;

    const int m_ld  = pixBase + lrowIdx;          // GEMM column (output pixel)
    const int n_ld  = m_ld / HWc;
    const int hw_ld = m_ld % HWc;
    const int h_ld  = hw_ld / W;
    const int w_ld  = hw_ld % W;

    uint4 wr0, wr1, ar0, ar1;

    // Branch-free halo handling: always load from a clamped (in-range)
    // address -> plain global_load_b128, then mask the data to zero for
    // out-of-bounds taps (v_cndmask/v_and instead of exec diamonds + scratch).
    auto g_load = [&](int kt) {
        const int rs = kt >> 1, half = kt & 1;
        const int r = rs / 3, s = rs % 3;
        const int c0 = half * 64;
        const signed char* wsrc =
            wq + (long)(ocBase + lrowIdx) * Kdim + kt * 64 + lchunk;
        wr0 = *(const uint4*)(wsrc);
        wr1 = *(const uint4*)(wsrc + 16);

        const int hh = h_ld + r - 1, ww = w_ld + s - 1;
        const bool ok = ((unsigned)hh < (unsigned)H) & ((unsigned)ww < (unsigned)W);
        const int hcl = min(max(hh, 0), H - 1);
        const int wcl = min(max(ww, 0), W - 1);
        const signed char* asrc =
            xq + (((long)n_ld * H + hcl) * W + wcl) * C + c0 + lchunk;
        uint4 a0 = *(const uint4*)(asrc);
        uint4 a1 = *(const uint4*)(asrc + 16);
        const unsigned msk = ok ? 0xFFFFFFFFu : 0u;
        ar0.x = a0.x & msk; ar0.y = a0.y & msk;
        ar0.z = a0.z & msk; ar0.w = a0.w & msk;
        ar1.x = a1.x & msk; ar1.y = a1.y & msk;
        ar1.z = a1.z & msk; ar1.w = a1.w & msk;
    };
    auto s_store = [&](int buf) {
        *(uint4*)&shW[buf][lrowIdx][lchunk]      = wr0;
        *(uint4*)&shW[buf][lrowIdx][lchunk + 16] = wr1;
        *(uint4*)&shA[buf][lrowIdx][lchunk]      = ar0;
        *(uint4*)&shA[buf][lrowIdx][lchunk + 16] = ar1;
    };

    // --- wave tiling: 2 waves along oc x 4 waves along pixels ---
    const int wv    = tid >> 5;
    const int lane  = tid & 31;
    const int lrow  = lane & 15;
    const int lhalf = lane >> 4;
    const int ocW   = (wv & 1) * 64;   // 4 A-frags of 16
    const int pxW   = (wv >> 1) * 32;  // 2 B-frags of 16

    // ISA 8-bit 16x64 operand layout: lane = row%16, K-interleave by half-wave:
    //   reg 2g+p <- bytes [g*16 + lhalf*8 + p*4 ...]
    auto frag = [&](const signed char* rowBase) {
        v8i f;
#pragma unroll
        for (int g = 0; g < 4; ++g) {
            int2 d = *(const int2*)(rowBase + g * 16 + lhalf * 8);
            f[2 * g]     = d.x;
            f[2 * g + 1] = d.y;
        }
        return f;
    };

    v8i acc[4][2];
    const v8i vzero = {0, 0, 0, 0, 0, 0, 0, 0};
#pragma unroll
    for (int i = 0; i < 4; ++i)
#pragma unroll
        for (int j = 0; j < 2; ++j) acc[i][j] = vzero;

    g_load(0);
    s_store(0);
    __syncthreads();

    for (int kt = 0; kt < KITER; ++kt) {
        const int cur = kt & 1;
        if (kt + 1 < KITER) g_load(kt + 1);        // prefetch next K-slab

        v8i bfr[2];
#pragma unroll
        for (int ni = 0; ni < 2; ++ni)
            bfr[ni] = frag(&shA[cur][pxW + ni * 16 + lrow][0]);

#pragma unroll
        for (int mi = 0; mi < 4; ++mi) {
            v8i afr = frag(&shW[cur][ocW + mi * 16 + lrow][0]);
#pragma unroll
            for (int ni = 0; ni < 2; ++ni)
                acc[mi][ni] = __builtin_amdgcn_wmma_i32_16x16x64_iu8(
                    /*sgn_a=*/true, afr, /*sgn_b=*/true, bfr[ni],
                    acc[mi][ni], /*reuse_a=*/false, /*reuse_b=*/false);
        }

        if (kt + 1 < KITER) s_store(cur ^ 1);
        __syncthreads();
    }

    // --- dequant + bias epilogue; lane dim = pixel -> coalesced stores ---
    const float sx    = fmaxf(wsf[0] / QMAX, 1e-8f);
    const float sw    = fmaxf(wsf[1] / QMAX, 1e-8f);
    const float scale = sx * sw;

#pragma unroll
    for (int mi = 0; mi < 4; ++mi) {
        const int ocb = ocBase + ocW + mi * 16 + lhalf * 8;  // + v
        float bb[8];
#pragma unroll
        for (int v = 0; v < 8; ++v) bb[v] = bias[ocb + v];
#pragma unroll
        for (int ni = 0; ni < 2; ++ni) {
            const int m     = pixBase + pxW + ni * 16 + lrow;
            const int n_img = m / HWc;
            const int hw    = m % HWc;
            float* ob = out + (long)n_img * OC * HWc + hw;
#pragma unroll
            for (int v = 0; v < 8; ++v)
                ob[(long)(ocb + v) * HWc] =
                    (float)acc[mi][ni][v] * scale + bb[v];
        }
    }
}

// ----------------------------------------------------------------- launch ---
extern "C" void kernel_launch(void* const* d_in, const int* in_sizes, int n_in,
                              void* d_out, int out_size, void* d_ws, size_t ws_size,
                              hipStream_t stream) {
    const float* x    = (const float*)d_in[0];
    const float* wgt  = (const float*)d_in[1];
    const float* bias = (const float*)d_in[2];
    float* out = (float*)d_out;

    // workspace: [0..1] absmax scalars | x_q int8 NHWC | w_q int8 [oc][K]
    float* wsf = (float*)d_ws;
    signed char* xq = (signed char*)d_ws + 256;
    signed char* wq = xq + (long)NB * HWc * C;      // +12,845,056 B
    // total ws use ~13.2 MB

    zero_scales_kernel<<<1, 64, 0, stream>>>(wsf);

    const long nx4 = (long)NB * C * HWc / 4;        // 3,211,264 float4
    absmax_kernel<<<2048, 256, 0, stream>>>((const float4*)x, nx4, wsf + 0);
    const long nw4 = (long)OC * C * 9 / 4;          // 73,728 float4
    absmax_kernel<<<288, 256, 0, stream>>>((const float4*)wgt, nw4, wsf + 1);

    const long qx_threads = (long)NB * HWc * (C / 4);
    quantize_x_kernel<<<(int)((qx_threads + 255) / 256), 256, 0, stream>>>(
        x, wsf, (char4*)xq);
    quantize_w_kernel<<<(OC * 9 * (C / 4) + 255) / 256, 256, 0, stream>>>(
        wgt, wsf, (char4*)wq);

    dim3 grid(Mpix / 128, OC / 128);                // 784 x 2
    conv_gemm_i8_wmma<<<grid, 256, 0, stream>>>(xq, wq, bias, wsf, out);
}
